// TopKLoRALinearSTE_69664369541281
// MI455X (gfx1250) — matrix operand order repaired and checked by reading, not compiled, not run on hardware
//
#include <hip/hip_runtime.h>
#include <hip/hip_bf16.h>

// ---------------------------------------------------------------------------
// TopK-LoRA-Linear-STE forward for MI455X (gfx1250, wave32, WMMA).
//   out = x @ W_base^T + b_base + 2.0 * (relu(x@A_w^T) * top16mask) @ B_w^T
// Pipeline:
//   1) cvt_bf16_kernel     : x, W_base, B_w -> bf16 in workspace (memory bound)
//   2) gate_kernel         : hg = 2 * relu(x@A_w^T) * top16mask   (bf16)
//   3) topk_lora_gemm      : bf16 WMMA GEMM, double-buffered LDS staged with
//                            global_load_async_to_lds_b128 (ASYNCcnt),
//                            fused LoRA tail + bias epilogue.
// ---------------------------------------------------------------------------

typedef __attribute__((ext_vector_type(16))) __bf16         v16bf;
typedef __attribute__((ext_vector_type(8)))  float          v8f;
typedef __attribute__((ext_vector_type(8)))  unsigned short v8u16;

#define D_IN   4096
#define D_OUT  4096
#define RLAT   64
#define KSEL   16
#define NROWS  8192          // 2 * 4096 rows of x
#define BK     64            // K slice per stage
#define LDSTR  72            // 64 bf16 + 8 pad -> 144B row stride, conflict-free

// f32 -> bf16, round-to-nearest-even
__device__ __forceinline__ unsigned short f2bf(float f) {
    unsigned u = __builtin_bit_cast(unsigned, f);
    u += 0x7FFFu + ((u >> 16) & 1u);
    return (unsigned short)(u >> 16);
}

__device__ __forceinline__ v8u16 pack8(float4 a, float4 b) {
    v8u16 p = { f2bf(a.x), f2bf(a.y), f2bf(a.z), f2bf(a.w),
                f2bf(b.x), f2bf(b.y), f2bf(b.z), f2bf(b.w) };
    return p;
}

__device__ __forceinline__ v16bf cat16(v8u16 lo, v8u16 hi) {
    return __builtin_bit_cast(v16bf,
        __builtin_shufflevector(lo, hi, 0,1,2,3,4,5,6,7,8,9,10,11,12,13,14,15));
}

// raw 32-bit LDS offset (addrspace(3) ptrtoint) for async-to-LDS ops
__device__ __forceinline__ unsigned lds_off(const void* p) {
    return (unsigned)(unsigned long long)
        (const __attribute__((address_space(3))) void*)p;
}

// one 16-byte global -> LDS async transfer per lane (ASYNCcnt-tracked)
__device__ __forceinline__ void async_b128(unsigned lds, const unsigned short* g) {
    unsigned long long ga = (unsigned long long)(const void*)g;
    asm volatile("global_load_async_to_lds_b128 %0, %1, off"
                 :: "v"(lds), "v"(ga) : "memory");
}

// ---------------------------------------------------------------------------
// Kernel 1: f32 -> bf16 bulk convert (8 elems / thread)
// ---------------------------------------------------------------------------
__global__ __launch_bounds__(256) void cvt_bf16_kernel(
    const float* __restrict__ src, unsigned short* __restrict__ dst, int n8) {
    int i = blockIdx.x * 256 + threadIdx.x;
    if (i < n8) {
        const float4* s = (const float4*)(src + (size_t)i * 8);
        float4 a = s[0], b = s[1];
        *(v8u16*)(dst + (size_t)i * 8) = pack8(a, b);
    }
}

// ---------------------------------------------------------------------------
// Kernel 2: gated latents.  One block (256 thr) per row of x.
// ---------------------------------------------------------------------------
__global__ __launch_bounds__(256) void gate_kernel(
    const float* __restrict__ x, const float* __restrict__ A_w,
    unsigned short* __restrict__ hg) {
    __shared__ float part[4][RLAT];
    __shared__ float zsh[RLAT];
    const int row   = blockIdx.x;
    const int tid   = threadIdx.x;
    const int r     = tid & (RLAT - 1);
    const int chunk = tid >> 6;                       // 0..3, 1024 elems each

    const float4* xr = (const float4*)(x   + (size_t)row * D_IN + chunk * 1024);
    const float4* ar = (const float4*)(A_w + (size_t)r   * D_IN + chunk * 1024);
    float s = 0.f;
#pragma unroll 4
    for (int i = 0; i < 256; ++i) {
        float4 a = xr[i], b = ar[i];
        s = fmaf(a.x, b.x, s); s = fmaf(a.y, b.y, s);
        s = fmaf(a.z, b.z, s); s = fmaf(a.w, b.w, s);
    }
    part[chunk][r] = s;
    __syncthreads();
    if (tid < RLAT) {
        float z = part[0][tid] + part[1][tid] + part[2][tid] + part[3][tid];
        zsh[tid] = z > 0.f ? z : 0.f;
    }
    __syncthreads();
    if (tid < RLAT) {
        float z = zsh[tid];
        int rank = 0;
#pragma unroll
        for (int j = 0; j < RLAT; ++j) {
            float zj = zsh[j];
            rank += (zj > z) || ((zj == z) && (j < tid));  // lax.top_k tie-break
        }
        float hv = (rank < KSEL) ? z * 2.0f : 0.f;          // SCALE folded in
        hg[(size_t)row * RLAT + tid] = f2bf(hv);
    }
}

// ---------------------------------------------------------------------------
// 8 WMMAs over one staged 32-wide K sub-slice (ks = 0 or 32 within the stage).
// A frag (16x32): lane 0-15 M=lane K={0..7,16..23}; lane 16-31 K={8..15,24..31}
// B frag (32x16): lanes 0-15 N=lane K={0..7,16..23}; lanes 16-31 N=lane-16 K={8..15,24..31}
// ---------------------------------------------------------------------------
__device__ __forceinline__ void mma_tiles(
    const unsigned short* __restrict__ Alds,
    const unsigned short* __restrict__ Blds,
    int ks, int wm, int wn, int lane, v8f (&acc)[4][2]) {
    const int mlo  = lane & 15;
    const int half = lane >> 4;
    v16bf bfrag[2];
#pragma unroll
    for (int tn = 0; tn < 2; ++tn) {
        const unsigned short* p =
            Blds + (wn * 32 + tn * 16 + mlo) * LDSTR + ks;
        bfrag[tn] = cat16(*(const v8u16*)(p + half * 8), *(const v8u16*)(p + 16 + half * 8));
    }
#pragma unroll
    for (int tm = 0; tm < 4; ++tm) {
        const unsigned short* p =
            Alds + (wm * 64 + tm * 16 + mlo) * LDSTR + ks;
        v16bf afrag = cat16(*(const v8u16*)(p + half * 8),
                            *(const v8u16*)(p + 16 + half * 8));
#pragma unroll
        for (int tn = 0; tn < 2; ++tn) {
            acc[tm][tn] = __builtin_amdgcn_wmma_f32_16x16x32_bf16(
                false, afrag, false, bfrag[tn], (short)0, acc[tm][tn], false, false);
        }
    }
}

// ---------------------------------------------------------------------------
// Kernel 3: fused base GEMM + LoRA up-projection + bias.
// Block tile 128x128, BK=64, 256 thr = 8 waves (2M x 4N), double-buffered LDS
// staged with global_load_async_to_lds_b128 (8 async b128 / wave / stage).
// ---------------------------------------------------------------------------
__global__ __launch_bounds__(256) void topk_lora_gemm_kernel(
    const unsigned short* __restrict__ xb,   // [8192,4096] bf16
    const unsigned short* __restrict__ Wb,   // [4096,4096] bf16
    const unsigned short* __restrict__ Bwb,  // [4096,  64] bf16
    const unsigned short* __restrict__ hg,   // [8192,  64] bf16 (gated, *2)
    const float* __restrict__ b_base, float* __restrict__ out) {
    __shared__ unsigned short Alds[2][128 * LDSTR];
    __shared__ unsigned short Blds[2][128 * LDSTR];

    const int tid    = threadIdx.x;
    const int lane   = tid & 31;
    const int wave   = tid >> 5;
    const int wm     = wave >> 2;              // 0..1 (M)
    const int wn     = wave & 3;               // 0..3 (N)
    const int rowblk = blockIdx.y * 128;
    const int colblk = blockIdx.x * 128;

    // staging decomposition: 2048 16B chunks per (A,B) tile pair, 4+4 per thread
    const int srow0 = (tid + 0)   >> 3, scc0 = (tid + 0)   & 7;
    const int srow1 = (tid + 256) >> 3, scc1 = (tid + 256) & 7;
    const int srow2 = (tid + 512) >> 3, scc2 = (tid + 512) & 7;
    const int srow3 = (tid + 768) >> 3, scc3 = (tid + 768) & 7;

    const unsigned aB[2] = { lds_off(&Alds[0][0]), lds_off(&Alds[1][0]) };
    const unsigned bB[2] = { lds_off(&Blds[0][0]), lds_off(&Blds[1][0]) };

    // issue one stage: A rows from gA (stride sA), B rows from gB (stride sB)
    auto stage = [&](int buf, const unsigned short* gA, int sA,
                     const unsigned short* gB, int sB) {
        async_b128(aB[buf] + srow0 * (LDSTR * 2) + scc0 * 16, gA + (size_t)srow0 * sA + scc0 * 8);
        async_b128(bB[buf] + srow0 * (LDSTR * 2) + scc0 * 16, gB + (size_t)srow0 * sB + scc0 * 8);
        async_b128(aB[buf] + srow1 * (LDSTR * 2) + scc1 * 16, gA + (size_t)srow1 * sA + scc1 * 8);
        async_b128(bB[buf] + srow1 * (LDSTR * 2) + scc1 * 16, gB + (size_t)srow1 * sB + scc1 * 8);
        async_b128(aB[buf] + srow2 * (LDSTR * 2) + scc2 * 16, gA + (size_t)srow2 * sA + scc2 * 8);
        async_b128(bB[buf] + srow2 * (LDSTR * 2) + scc2 * 16, gB + (size_t)srow2 * sB + scc2 * 8);
        async_b128(aB[buf] + srow3 * (LDSTR * 2) + scc3 * 16, gA + (size_t)srow3 * sA + scc3 * 8);
        async_b128(bB[buf] + srow3 * (LDSTR * 2) + scc3 * 16, gB + (size_t)srow3 * sB + scc3 * 8);
    };

    v8f acc[4][2];
    v8f zero = {};
#pragma unroll
    for (int tm = 0; tm < 4; ++tm)
#pragma unroll
        for (int tn = 0; tn < 2; ++tn) acc[tm][tn] = zero;

    const unsigned short* gx = xb + (size_t)rowblk * D_IN;
    const unsigned short* gw = Wb + (size_t)colblk * D_IN;

    // ---- base GEMM: 64 double-buffered stages of BK=64 ----
    stage(0, gx, D_IN, gw, D_IN);
    const int NSTAGE = D_IN / BK;
    for (int s = 0; s < NSTAGE; ++s) {
        const int buf = s & 1;
        if (s + 1 < NSTAGE) {
            stage(buf ^ 1, gx + (s + 1) * BK, D_IN, gw + (s + 1) * BK, D_IN);
            asm volatile("s_wait_asynccnt 0x8" ::: "memory");  // stage s landed
        } else {
            asm volatile("s_wait_asynccnt 0x0" ::: "memory");
        }
        __syncthreads();
        mma_tiles(&Alds[buf][0], &Blds[buf][0],  0, wm, wn, lane, acc);
        mma_tiles(&Alds[buf][0], &Blds[buf][0], 32, wm, wn, lane, acc);
        __syncthreads();                       // buf reusable for stage s+2
    }

    // ---- LoRA up-projection: one BK=64 stage, same accumulators ----
    stage(0, hg + (size_t)rowblk * RLAT, RLAT, Bwb + (size_t)colblk * RLAT, RLAT);
    asm volatile("s_wait_asynccnt 0x0" ::: "memory");
    __syncthreads();
    mma_tiles(&Alds[0][0], &Blds[0][0],  0, wm, wn, lane, acc);
    mma_tiles(&Alds[0][0], &Blds[0][0], 32, wm, wn, lane, acc);

    // ---- epilogue: + b_base ; C/D layout row = base + v + 8*(lane>=16) ----
    const int mlo  = lane & 15;
    const int half = lane >> 4;
#pragma unroll
    for (int tn = 0; tn < 2; ++tn) {
        const int col = colblk + wn * 32 + tn * 16 + mlo;
        const float bb = b_base[col];
#pragma unroll
        for (int tm = 0; tm < 4; ++tm) {
            const int row = rowblk + wm * 64 + tm * 16 + half * 8;
#pragma unroll
            for (int v = 0; v < 8; ++v) {
                out[(size_t)(row + v) * D_OUT + col] = acc[tm][tn][v] + bb;
            }
        }
    }
}

// ---------------------------------------------------------------------------
extern "C" void kernel_launch(void* const* d_in, const int* in_sizes, int n_in,
                              void* d_out, int out_size, void* d_ws, size_t ws_size,
                              hipStream_t stream) {
    (void)in_sizes; (void)n_in; (void)out_size; (void)ws_size;
    const float* x      = (const float*)d_in[0];
    const float* W_base = (const float*)d_in[1];
    const float* b_base = (const float*)d_in[2];
    const float* A_w    = (const float*)d_in[3];
    const float* B_w    = (const float*)d_in[4];
    float* out = (float*)d_out;

    // workspace layout (bf16 halves), all 16B aligned
    unsigned short* xb  = (unsigned short*)d_ws;            // 64 MiB
    unsigned short* Wb  = xb  + (size_t)NROWS * D_IN;       // 32 MiB
    unsigned short* Bwb = Wb  + (size_t)D_OUT * D_IN;       // 0.5 MiB
    unsigned short* hg  = Bwb + (size_t)D_OUT * RLAT;       // 1 MiB

    const int nx = NROWS * D_IN / 8, nw = D_OUT * D_IN / 8, nb = D_OUT * RLAT / 8;
    cvt_bf16_kernel<<<(nx + 255) / 256, 256, 0, stream>>>(x, xb, nx);
    cvt_bf16_kernel<<<(nw + 255) / 256, 256, 0, stream>>>(W_base, Wb, nw);
    cvt_bf16_kernel<<<(nb + 255) / 256, 256, 0, stream>>>(B_w, Bwb, nb);

    gate_kernel<<<NROWS, 256, 0, stream>>>(x, A_w, hg);

    dim3 grid(D_OUT / 128, NROWS / 128);                    // (32, 64)
    topk_lora_gemm_kernel<<<grid, 256, 0, stream>>>(xb, Wb, Bwb, hg, b_base, out);
}
